// ConformerEncoderLayer_50989851738750
// MI455X (gfx1250) — matrix operand order repaired and checked by reading, not compiled, not run on hardware
//
#include <hip/hip_runtime.h>

// ---------------------------------------------------------------------------
// Conformer encoder layer forward for MI455X (gfx1250, wave32, WMMA).
// All GEMMs go through v_wmma_f32_16x16x32_f16 (fp32 -> f16 operands,
// f32 accumulate). Every GEMM uses the contiguous "weight-like" B layout
// (v is pre-transposed in prep), so all fragment loads are b128.
// Each wave computes a 32(M) x 64(N) tile: 8 WMMAs per 32-wide K step.
// ---------------------------------------------------------------------------

namespace {
constexpr int TT  = 512;        // sequence length
constexpr int NB  = 16;         // batch
constexpr int EE  = 512;        // model dim
constexpr int HH  = 8;          // heads
constexpr int DFF = 2048;       // feed-forward dim
constexpr int KK  = 31;         // depthwise conv kernel
constexpr int DD  = EE / HH;    // 64 head dim
constexpr int ROWS = TT * NB;   // 8192 token rows
}

typedef __attribute__((ext_vector_type(16))) _Float16 v16h;
typedef __attribute__((ext_vector_type(8)))  float    v8f;

__device__ __forceinline__ float sigmoidf_(float x) {
  return 1.0f / (1.0f + __expf(-x));
}
__device__ __forceinline__ float dswishf_(float x) {
  // x * sigmoid(x - 1)
  return x / (1.0f + __expf(1.0f - x));
}
__device__ __forceinline__ void cvt4_(v16h& dst, int base, float4 f) {
  dst[base + 0] = (_Float16)f.x;
  dst[base + 1] = (_Float16)f.y;
  dst[base + 2] = (_Float16)f.z;
  dst[base + 3] = (_Float16)f.w;
}

// ---------------------------------------------------------------------------
// Batched GEMM:  C[b] = act(A[b] @ B[b]^T + bias) (+ res)
//   A: (M x K) row-major, leading dim lda
//   B: (Ncols x K) row-major, leading dim ldb  (computes x @ W^T)
//   batch offset for X = (b / binner) * sX1 + (b % binner) * sX2
// 4 waves/block; wave computes 32(M) x 64(N): 2 A-frags, 4 B-frags, 8 WMMAs
// per 32-wide K step. Fragment layouts per CDNA5 ISA 7.12.2.
// ---------------------------------------------------------------------------
__global__ __launch_bounds__(128)
void wmma_gemm_f16(const float* __restrict__ A, const float* __restrict__ B,
                   const float* __restrict__ bias, const float* __restrict__ res,
                   float* __restrict__ C,
                   int M, int Nc, int Kd,
                   int lda, int ldb, int ldc,
                   int binner,
                   long long sA1, long long sA2,
                   long long sB1, long long sB2,
                   long long sC1, long long sC2,
                   int act)
{
  const int wave = threadIdx.x >> 5;
  const int lane = threadIdx.x & 31;
  const int half = lane >> 4;      // 0: lanes 0-15, 1: lanes 16-31
  const int r    = lane & 15;

  const int b = blockIdx.z;
  const long long offA = (long long)(b / binner) * sA1 + (long long)(b % binner) * sA2;
  const long long offB = (long long)(b / binner) * sB1 + (long long)(b % binner) * sB2;
  const long long offC = (long long)(b / binner) * sC1 + (long long)(b % binner) * sC2;

  const int m0 = blockIdx.y * 128 + wave * 32;
  const int n0 = blockIdx.x * 64;
  if (m0 >= M) return;                       // wave-uniform

  // A row bases for the two 16-row subtiles (clamped; OOB rows masked at store)
  const int mr0 = (m0 + r      < M) ? (m0 + r)      : (M - 1);
  const int mr1 = (m0 + 16 + r < M) ? (m0 + 16 + r) : (M - 1);
  const float* a0p = A + offA + (long long)mr0 * lda + half * 8;
  const float* a1p = A + offA + (long long)mr1 * lda + half * 8;

  // B column bases for the four 16-col subtiles
  const float* bp[4];
#pragma unroll
  for (int nt = 0; nt < 4; ++nt) {
    int ncol = n0 + nt * 16 + r;
    int ncl  = (ncol < Nc) ? ncol : (Nc - 1);
    bp[nt] = B + offB + (long long)ncl * ldb + half * 8;
  }

  v8f acc[2][4] = { { v8f{}, v8f{}, v8f{}, v8f{} },
                    { v8f{}, v8f{}, v8f{}, v8f{} } };

  for (int kk = 0; kk < Kd; kk += 32) {
    if (kk + 32 < Kd) {
      __builtin_prefetch(a0p + kk + 32, 0, 1);
      __builtin_prefetch(bp[0] + kk + 32, 0, 1);
      __builtin_prefetch(bp[2] + kk + 32, 0, 1);
    }
    // A fragments: 16x32 f16. half 0: K = kk+[0..7], kk+[16..23]; half 1: +8.
    v16h af[2];
    {
      const float4* p = (const float4*)(a0p + kk);
      cvt4_(af[0], 0, p[0]); cvt4_(af[0], 4, p[1]);
      cvt4_(af[0], 8, p[4]); cvt4_(af[0], 12, p[5]);
      const float4* q = (const float4*)(a1p + kk);
      cvt4_(af[1], 0, q[0]); cvt4_(af[1], 4, q[1]);
      cvt4_(af[1], 8, q[4]); cvt4_(af[1], 12, q[5]);
    }
    // B fragments: 32x16 f16. Lane r = column; half 0: K = kk+[0..7], kk+[16..23]; half 1: +8.
    v16h bf[4];
#pragma unroll
    for (int nt = 0; nt < 4; ++nt) {
      const float4* p = (const float4*)(bp[nt] + kk);
      cvt4_(bf[nt], 0, p[0]); cvt4_(bf[nt], 4, p[1]);
      cvt4_(bf[nt], 8, p[4]); cvt4_(bf[nt], 12, p[5]);
    }
#pragma unroll
    for (int mt = 0; mt < 2; ++mt)
#pragma unroll
      for (int nt = 0; nt < 4; ++nt)
        acc[mt][nt] = __builtin_amdgcn_wmma_f32_16x16x32_f16(
            false, af[mt], false, bf[nt], (short)0, acc[mt][nt], false, false);
  }

  // Epilogue. C layout: lane r = column, VGPR v = row v + 8*half.
#pragma unroll
  for (int nt = 0; nt < 4; ++nt) {
    int ncol = n0 + nt * 16 + r;
    if (ncol >= Nc) continue;
    float bv = bias ? bias[ncol] : 0.0f;
#pragma unroll
    for (int mt = 0; mt < 2; ++mt) {
#pragma unroll
      for (int v = 0; v < 8; ++v) {
        int m = m0 + mt * 16 + v + half * 8;
        if (m >= M) continue;
        float val = acc[mt][nt][v] + bv;
        if (act == 1) val = dswishf_(val);
        long long ci = offC + (long long)m * ldc + ncol;
        if (res) val += res[ci];
        C[ci] = val;
      }
    }
  }
}

// ---------------------------------------------------------------------------
// qkv (ROWS x 3E, row = t*NB+n) -> qU/qV/kT (N,H,T,D) and vT (N,H,D,T).
// qU = q*D^-0.5 + bias_u[h,:], qV = q*D^-0.5 + bias_v[h,:].
// v is transposed so attn@v can use the contiguous-K B loader.
// ---------------------------------------------------------------------------
__global__ __launch_bounds__(256)
void prep_qkv(const float* __restrict__ qkv,
              const float* __restrict__ bu, const float* __restrict__ bv,
              float* __restrict__ qU, float* __restrict__ qV,
              float* __restrict__ kT, float* __restrict__ vT)
{
  long long i = (long long)blockIdx.x * 256 + threadIdx.x;
  const long long total = (long long)NB * HH * TT * DD;
  if (i >= total) return;
  int d = (int)(i % DD);
  long long t1 = i / DD;
  int t = (int)(t1 % TT);
  long long t2 = t1 / TT;     // batch index n*H + h
  int h = (int)(t2 % HH);
  long long base = ((long long)t * NB + (int)(t2 / HH)) * (3 * EE);
  const float scale = 0.125f;  // D^-0.5, D = 64
  float q = qkv[base + h * DD + d] * scale;
  qU[i] = q + bu[h * DD + d];
  qV[i] = q + bv[h * DD + d];
  kT[i] = qkv[base + EE + h * DD + d];
  vT[(t2 * DD + d) * TT + t] = qkv[base + 2 * EE + h * DD + d];
}

// ---------------------------------------------------------------------------
// Fused rel-shift + softmax. One block per (batch,row):
//   attn[t,s] = softmax_s( ac[t,s] + bd[t, (T-1)+s-t] ), written in place.
// bd rows padded to stride 1024.
// ---------------------------------------------------------------------------
__global__ __launch_bounds__(256)
void attn_softmax(float* __restrict__ scores, const float* __restrict__ bd)
{
  long long rowi = blockIdx.x;           // over N*H*T
  int t = (int)(rowi % TT);
  long long bb = rowi / TT;
  float* sc = scores + rowi * TT;
  const float* bdr = bd + (bb * TT + t) * 1024LL;
  __shared__ float red[256];
  int tid = threadIdx.x;
  float v0 = sc[tid]       + bdr[(TT - 1) + tid - t];
  float v1 = sc[tid + 256] + bdr[(TT - 1) + tid + 256 - t];
  red[tid] = fmaxf(v0, v1);
  __syncthreads();
  for (int s = 128; s > 0; s >>= 1) {
    if (tid < s) red[tid] = fmaxf(red[tid], red[tid + s]);
    __syncthreads();
  }
  float mx = red[0];
  __syncthreads();
  float e0 = __expf(v0 - mx), e1 = __expf(v1 - mx);
  red[tid] = e0 + e1;
  __syncthreads();
  for (int s = 128; s > 0; s >>= 1) {
    if (tid < s) red[tid] += red[tid + s];
    __syncthreads();
  }
  float inv = 1.0f / red[0];
  sc[tid] = e0 * inv;
  sc[tid + 256] = e1 * inv;
}

// GLU over channel dim: g[row,c] = z[row,c] * sigmoid(z[row,E+c])
__global__ __launch_bounds__(256)
void glu_kernel(const float* __restrict__ z, float* __restrict__ g)
{
  long long i = (long long)blockIdx.x * 256 + threadIdx.x;   // T*N*E
  if (i >= (long long)ROWS * EE) return;
  int c = (int)(i % EE);
  long long row = i / EE;
  float a = z[row * (2 * EE) + c];
  float b = z[row * (2 * EE) + EE + c];
  g[i] = a * sigmoidf_(b);
}

// Depthwise conv over t (K=31, same padding) + DoubleSwish.
__global__ __launch_bounds__(256)
void dwconv_dswish(const float* __restrict__ g, const float* __restrict__ w,
                   const float* __restrict__ b, float* __restrict__ out)
{
  long long i = (long long)blockIdx.x * 256 + threadIdx.x;   // (t*N+n)*E + c
  if (i >= (long long)ROWS * EE) return;
  int c = (int)(i % EE);
  long long tn = i / EE;
  int n = (int)(tn % NB);
  int t = (int)(tn / NB);
  float acc = b[c];
#pragma unroll
  for (int j = 0; j < KK; ++j) {
    int ts = t + j - (KK - 1) / 2;
    if (ts >= 0 && ts < TT)
      acc += g[((long long)ts * NB + n) * EE + c] * w[c * KK + j];
  }
  out[i] = dswishf_(acc);
}

// BasicNorm: out = x * (mean(x^2) + exp(eps))^-0.5 per row of E=512.
__global__ __launch_bounds__(256)
void basic_norm(const float* __restrict__ x, const float* __restrict__ eps_p,
                float* __restrict__ out)
{
  long long row = blockIdx.x;            // over T*N
  const float* xr = x + row * EE;
  __shared__ float red[256];
  int tid = threadIdx.x;
  float a = xr[tid], b = xr[tid + 256];
  red[tid] = a * a + b * b;
  __syncthreads();
  for (int s = 128; s > 0; s >>= 1) {
    if (tid < s) red[tid] += red[tid + s];
    __syncthreads();
  }
  float scale = rsqrtf(red[0] * (1.0f / EE) + __expf(eps_p[0]));
  out[row * EE + tid] = a * scale;
  out[row * EE + tid + 256] = b * scale;
}

// ---------------------------------------------------------------------------
static inline void launch_gemm(hipStream_t s,
    const float* A, const float* B, const float* bias, const float* res,
    float* C, int M, int Nc, int Kd, int lda, int ldb, int ldc,
    int batches, int binner,
    long long sA1, long long sA2, long long sB1, long long sB2,
    long long sC1, long long sC2, int act)
{
  dim3 grid((Nc + 63) / 64, (M + 127) / 128, batches);
  wmma_gemm_f16<<<grid, 128, 0, s>>>(A, B, bias, res, C, M, Nc, Kd,
                                     lda, ldb, ldc, binner,
                                     sA1, sA2, sB1, sB2, sC1, sC2, act);
}

extern "C" void kernel_launch(void* const* d_in, const int* in_sizes, int n_in,
                              void* d_out, int out_size, void* d_ws, size_t ws_size,
                              hipStream_t stream) {
  (void)in_sizes; (void)n_in; (void)out_size; (void)ws_size;

  const float* src        = (const float*)d_in[0];
  const float* pos_emb    = (const float*)d_in[1];   // (1023, 512)
  const float* ffm_w1     = (const float*)d_in[2];
  const float* ffm_b1     = (const float*)d_in[3];
  const float* ffm_w2     = (const float*)d_in[4];
  const float* ffm_b2     = (const float*)d_in[5];
  const float* in_proj_w  = (const float*)d_in[6];
  const float* in_proj_b  = (const float*)d_in[7];
  const float* pos_w      = (const float*)d_in[8];
  const float* pos_bias_u = (const float*)d_in[9];
  const float* pos_bias_v = (const float*)d_in[10];
  const float* out_w      = (const float*)d_in[11];
  const float* out_b      = (const float*)d_in[12];
  const float* cpw1_w     = (const float*)d_in[13];
  const float* cpw1_b     = (const float*)d_in[14];
  const float* cdw_w      = (const float*)d_in[15];
  const float* cdw_b      = (const float*)d_in[16];
  const float* cpw2_w     = (const float*)d_in[17];
  const float* cpw2_b     = (const float*)d_in[18];
  const float* ff_w1      = (const float*)d_in[19];
  const float* ff_b1      = (const float*)d_in[20];
  const float* ff_w2      = (const float*)d_in[21];
  const float* ff_b2      = (const float*)d_in[22];
  const float* norm_eps   = (const float*)d_in[23];
  float* out = (float*)d_out;

  // Workspace layout (floats). Regions reused across phases.
  float* ws = (float*)d_ws;
  float* hbuf   = ws;                         // 16.78M: ff hidden / conv z
  float* xA     = ws + 16777216LL;            // 4.19M
  float* xB     = ws + 20971520LL;            // 4.19M
  float* qkvb   = ws + 25165824LL;            // 12.58M (later attn_out/g/dconv)
  float* attn_o = qkvb;                       // 4.19M (qkv dead after prep)
  float* gbuf   = qkvb + 4194304LL;           // 4.19M
  float* dbuf   = qkvb + 8388608LL;           // 4.19M
  float* qU     = ws + 37748736LL;            // 4.19M each
  float* qV     = ws + 41943040LL;
  float* kT     = ws + 46137344LL;
  float* vT     = ws + 50331648LL;            // (N,H,D,T)
  float* scores = ws + 54525952LL;            // 33.55M (N,H,T,T)
  float* bdbuf  = ws + 88080384LL;            // 67.11M (N,H,T,1024 padded)
  float* pbuf   = ws + 155189248LL;           // 0.52M  (1023,512)

  const long long sTD = (long long)TT * DD;          // 32768
  const long long sHTD = (long long)HH * sTD;        // 262144
  const long long sTT = (long long)TT * TT;          // 262144
  const long long sHTT = (long long)HH * sTT;        // 2097152

  // 1) Macaron FFN: h = dswish(src @ w1^T + b1)
  launch_gemm(stream, src, ffm_w1, ffm_b1, nullptr, hbuf,
              ROWS, DFF, EE, EE, EE, DFF, 1, 1, 0,0,0,0,0,0, 1);
  //    x1 = src + h @ w2^T + b2
  launch_gemm(stream, hbuf, ffm_w2, ffm_b2, src, xA,
              ROWS, EE, DFF, DFF, DFF, EE, 1, 1, 0,0,0,0,0,0, 0);

  // 2) qkv = x1 @ in_proj_w^T + b
  launch_gemm(stream, xA, in_proj_w, in_proj_b, nullptr, qkvb,
              ROWS, 3 * EE, EE, EE, EE, 3 * EE, 1, 1, 0,0,0,0,0,0, 0);
  //    p = pos_emb @ pos_w^T   (M = 1023 handled by clamps/guards)
  launch_gemm(stream, pos_emb, pos_w, nullptr, nullptr, pbuf,
              2 * TT - 1, EE, EE, EE, EE, EE, 1, 1, 0,0,0,0,0,0, 0);

  // 3) split/scale/bias into (N,H,T,D); v transposed to (N,H,D,T)
  prep_qkv<<<(NB * HH * TT * DD + 255) / 256, 256, 0, stream>>>(
      qkvb, pos_bias_u, pos_bias_v, qU, qV, kT, vT);

  // 4) ac[b] = qU[b] @ kT[b]^T  -> (T,T) per (n,h)
  launch_gemm(stream, qU, kT, nullptr, nullptr, scores,
              TT, TT, DD, DD, DD, TT, NB * HH, HH,
              sHTD, sTD, sHTD, sTD, sHTT, sTT, 0);
  // 5) bd[b] = qV[b] @ p[h]^T  -> (T,1023), ldc padded to 1024
  launch_gemm(stream, qV, pbuf, nullptr, nullptr, bdbuf,
              TT, 2 * TT - 1, DD, DD, EE, 1024, NB * HH, HH,
              sHTD, sTD, 0, DD, (long long)HH * TT * 1024, (long long)TT * 1024,
              0);

  // 6) rel-shift + softmax (in place on scores)
  attn_softmax<<<NB * HH * TT, 256, 0, stream>>>(scores, bdbuf);

  // 7) o[b] = attn[b] @ vT[b]^T (vT is (D,T): contiguous-K B operand),
  //    scattered into (T, N*E) token layout
  launch_gemm(stream, scores, vT, nullptr, nullptr, attn_o,
              TT, DD, TT, TT, TT, NB * EE, NB * HH, HH,
              sHTT, sTT, sHTD, sTD, (long long)EE, (long long)DD, 0);

  // 8) x2 = x1 + o @ out_w^T + out_b
  launch_gemm(stream, attn_o, out_w, out_b, xA, xB,
              ROWS, EE, EE, EE, EE, EE, 1, 1, 0,0,0,0,0,0, 0);

  // 9) conv module: z = x2 @ pw1^T + b  (reuse hbuf)
  launch_gemm(stream, xB, cpw1_w, cpw1_b, nullptr, hbuf,
              ROWS, 2 * EE, EE, EE, EE, 2 * EE, 1, 1, 0,0,0,0,0,0, 0);
  glu_kernel<<<(ROWS * EE + 255) / 256, 256, 0, stream>>>(hbuf, gbuf);
  dwconv_dswish<<<(ROWS * EE + 255) / 256, 256, 0, stream>>>(gbuf, cdw_w, cdw_b, dbuf);
  //    x3 = x2 + d @ pw2^T + b
  launch_gemm(stream, dbuf, cpw2_w, cpw2_b, xB, xA,
              ROWS, EE, EE, EE, EE, EE, 1, 1, 0,0,0,0,0,0, 0);

  // 10) second FFN
  launch_gemm(stream, xA, ff_w1, ff_b1, nullptr, hbuf,
              ROWS, DFF, EE, EE, EE, DFF, 1, 1, 0,0,0,0,0,0, 1);
  launch_gemm(stream, hbuf, ff_w2, ff_b2, xA, xB,
              ROWS, EE, DFF, DFF, DFF, EE, 1, 1, 0,0,0,0,0,0, 0);

  // 11) BasicNorm -> output
  basic_norm<<<ROWS, 256, 0, stream>>>(xB, norm_eps, out);
}
